// GNCA_85487029059747
// MI455X (gfx1250) — matrix operand order repaired and checked by reading, not compiled, not run on hardware
//
#include <hip/hip_runtime.h>
#include <hip/hip_bf16.h>

typedef __attribute__((ext_vector_type(16))) _Float16 v16h;
typedef __attribute__((ext_vector_type(8)))  float    v8f;

#define B_ 8
#define M_ 3072
#define C_ 10
#define H_ 128
#define K_ 32
#define RADIUS_    0.04f
#define ACC_SCALE_ 0.02f
#define MAX_VEL_   0.02f
#define NOISE_     0.002f

// ---------------------------------------------------------------------------
// Kernel 1: brute-force periodic KNN, sorted top-K=32 by distance per particle
// One thread per particle; positions tiled through LDS; per-thread sorted list
// kept in LDS laid out [entry][thread] so shifts are bank-conflict-free.
// ---------------------------------------------------------------------------
__global__ __launch_bounds__(128) void knn_kernel(const float* __restrict__ x,
                                                  int*   __restrict__ nbr_idx,
                                                  float* __restrict__ nbr_dist,
                                                  int*   __restrict__ nbr_cnt) {
  __shared__ float sd[K_][128];
  __shared__ int   si[K_][128];
  __shared__ float spx[128];
  __shared__ float spy[128];

  const int t  = threadIdx.x;
  const int gp = blockIdx.x * 128 + t;          // global particle (b*M + i)
  const int b  = gp / M_;
  const int i  = gp - b * M_;
  const float pix = x[(size_t)gp * C_ + 0];
  const float piy = x[(size_t)gp * C_ + 1];
  const int base = b * M_;
  int cnt = 0;

  for (int tile = 0; tile < M_ / 128; ++tile) {
    __syncthreads();
    const int jl = tile * 128 + t;
    spx[t] = x[(size_t)(base + jl) * C_ + 0];
    spy[t] = x[(size_t)(base + jl) * C_ + 1];
    __syncthreads();
    for (int jj = 0; jj < 128; ++jj) {
      const int j2 = tile * 128 + jj;
      float dx = pix - spx[jj]; dx -= rintf(dx);   // periodic wrap (round-half-even)
      float dy = piy - spy[jj]; dy -= rintf(dy);
      const float d = sqrtf(dx * dx + dy * dy + 1e-12f);
      if (d < RADIUS_ && j2 != i) {
        if (cnt == K_ && d >= sd[K_ - 1][t]) continue;
        int p = (cnt < K_) ? cnt : (K_ - 1);
        while (p > 0 && sd[p - 1][t] > d) {        // stable: equal keeps earlier idx
          sd[p][t] = sd[p - 1][t];
          si[p][t] = si[p - 1][t];
          --p;
        }
        sd[p][t] = d;
        si[p][t] = j2;
        if (cnt < K_) ++cnt;
      }
    }
  }
  nbr_cnt[gp] = cnt;
  for (int k = 0; k < K_; ++k) {
    nbr_idx[(size_t)gp * K_ + k]  = (k < cnt) ? si[k][t] : 0;
    nbr_dist[(size_t)gp * K_ + k] = (k < cnt) ? sd[k][t] : 0.f;
  }
}

// ---------------------------------------------------------------------------
// Kernel 2: fused edge-MLP (WMMA f16 16x16x32) + aggregation + layer2 + physics
// One wave32 per particle per iteration; 4 particles per wave; W1 held as
// pre-formatted WMMA B tiles in registers for the whole kernel.
//
// Validity masking trick: invalid edge rows carry exactly-zero A features, so
// their WMMA output is exactly 0.0 and their post-bias ReLU value is the
// constant relu(b1[h]). Hence
//     sum_valid relu(c + b) = sum_all32 relu(c + b) - (32 - cnt) * relu(b)
// which replaces 16 v_cmp+v_cndmask per N-tile with one FMA correction.
// ---------------------------------------------------------------------------
__global__ __launch_bounds__(128) void gnca_mlp_kernel(
    const float* __restrict__ x,  const float* __restrict__ W1,
    const float* __restrict__ b1, const float* __restrict__ W2,
    const float* __restrict__ b2, const float* __restrict__ noise_u,
    const int* __restrict__ nbr_idx, const float* __restrict__ nbr_dist,
    const int* __restrict__ nbr_cnt, float* __restrict__ out) {
  __shared__ _Float16 w1h[32 * H_];       // W1 padded to K=32 rows, [f][h]
  __shared__ float    b1s[H_];
  __shared__ float    w2c0[H_], w2c1[H_]; // only cols 0,1 of W2 feed the state
  __shared__ _Float16 Alds[4][K_][32];    // per-wave 32x32 f16 A staging
  __shared__ float    agg[4][H_];         // per-wave aggregated hidden vector

  const int t    = threadIdx.x;
  const int lane = t & 31;
  const int wave = t >> 5;

  // ---- stage shared weights ----
  {
    const int h = t;                      // 0..127
    for (int f = 0; f < 12; ++f)  w1h[f * H_ + h] = (_Float16)W1[f * H_ + h];
    for (int f = 12; f < 32; ++f) w1h[f * H_ + h] = (_Float16)0.f;
    b1s[h]  = b1[h];
    w2c0[h] = W2[h * 7 + 0];
    w2c1[h] = W2[h * 7 + 1];
  }
  __syncthreads();

  // ---- resident B tiles in WMMA B layout (16-bit 32x16):
  // lane holds column N = lane%16; K rows 0..7 & 16..23 (lanes<16) / 8..15 & 24..31 (lanes>=16)
  v16h Breg[8];
  {
    const int col = lane & 15;
    const int kb  = (lane < 16) ? 0 : 16;
    for (int n = 0; n < 8; ++n)
      for (int e = 0; e < 16; ++e)
        Breg[n][e] = w1h[(((e & 8) << 1) + (kb >> 1) + (e & 7)) * H_ + (n * 16 + col)];
  }

  // ---- hoisted per-lane constants (loop-invariant across particles) ----
  const int col = lane & 15;
  float biasr[8], rbias[8];               // b1 per N-tile column, and relu(b1)
  for (int n = 0; n < 8; ++n) {
    biasr[n] = b1s[n * 16 + col];
    rbias[n] = biasr[n] > 0.f ? biasr[n] : 0.f;
  }
  float w2a[4], w2b[4];                   // layer-2 weight slices per lane
  for (int tt = 0; tt < 4; ++tt) {
    w2a[tt] = w2c0[lane + 32 * tt];
    w2b[tt] = w2c1[lane + 32 * tt];
  }
  const float b20 = b2[0], b21 = b2[1];

  const int wgid = blockIdx.x * 4 + wave;        // 0..6143
  for (int it = 0; it < 4; ++it) {
    const int gp = wgid + it * 6144;             // 0..24575, uniform trip count
    const int b  = gp / M_;
    const float* xi = x + (size_t)gp * C_;
    const float pix = xi[0], piy = xi[1];
    const int cnt = nbr_cnt[gp];
    const float ninval = (float)(K_ - cnt);      // # of zero-feature rows

    // ---- build A row for edge k = lane (12 features, zero-padded to 32) ----
    bool ncell = false;
    __syncthreads();                             // Alds reuse across iterations
    {
      const bool valid = lane < cnt;
      float f[12];
      if (valid) {
        const int   j = nbr_idx[(size_t)gp * K_ + lane];
        const float d = nbr_dist[(size_t)gp * K_ + lane];
        const float* xj = x + (size_t)(b * M_ + j) * C_;
        float dvx = pix - xj[0]; dvx -= rintf(dvx);
        float dvy = piy - xj[1]; dvy -= rintf(dvy);
        for (int c = 0; c < 8; ++c) f[c] = xj[2 + c];  // xj channels 2..9
        f[8] = d; f[9] = dvx; f[10] = dvy; f[11] = xj[4];
        ncell = (xj[4] == 1.0f);
      } else {
        for (int c = 0; c < 12; ++c) f[c] = 0.f;       // exact-zero rows
      }
      for (int c = 0; c < 12; ++c)  Alds[wave][lane][c] = (_Float16)f[c];
      for (int c = 12; c < 32; ++c) Alds[wave][lane][c] = (_Float16)0.f;
    }
    __syncthreads();

    // ---- read A tiles in WMMA A layout (16-bit 16x32):
    // lane<16: row=lane, K 0..7 & 16..23 ; lane>=16: row=lane-16, K 8..15 & 24..31
    v16h a0, a1;
    {
      const int row = lane & 15;
      const int hs  = (lane < 16) ? 0 : 8;
      for (int e = 0; e < 8; ++e) {
        a0[e]     = Alds[wave][row][hs + e];
        a0[8 + e] = Alds[wave][row][16 + hs + e];
        a1[e]     = Alds[wave][16 + row][hs + e];
        a1[8 + e] = Alds[wave][16 + row][16 + hs + e];
      }
    }

    // ---- layer 1: 16 WMMAs; unmasked bias+ReLU sum, then closed-form fix ----
    for (int n = 0; n < 8; ++n) {
      v8f cz = {};
      v8f c0 = __builtin_amdgcn_wmma_f32_16x16x32_f16(false, a0, false, Breg[n],
                                                      (short)0, cz, false, false);
      v8f c1 = __builtin_amdgcn_wmma_f32_16x16x32_f16(false, a1, false, Breg[n],
                                                      (short)0, cz, false, false);
      const float bias = biasr[n];
      float part = 0.f;
      for (int r = 0; r < 8; ++r) {
        float v0 = c0[r] + bias; part += (v0 > 0.f ? v0 : 0.f);
        float v1 = c1[r] + bias; part += (v1 > 0.f ? v1 : 0.f);
      }
      part += __shfl_xor(part, 16);              // fold the two half-lanes
      if (lane < 16)
        agg[wave][n * 16 + col] = part - ninval * rbias[n];
    }
    __builtin_amdgcn_wave_barrier();             // same-wave LDS RAW order

    // ---- layer 2: only output columns 0,1 are consumed by the update ----
    float acc0 = 0.f, acc1 = 0.f;
    for (int tt = 0; tt < 4; ++tt) {
      const float a = agg[wave][lane + 32 * tt];
      acc0 += a * w2a[tt];
      acc1 += a * w2b[tt];
    }
    for (int off = 16; off >= 1; off >>= 1) {
      acc0 += __shfl_xor(acc0, off);
      acc1 += __shfl_xor(acc1, off);
    }

    // ---- deg_cell via wave32 ballot over the 32 edge lanes ----
    const unsigned long long bal = __ballot(ncell);
    const int deg = __popcll(bal);

    // ---- physics update (uniform across lanes) ----
    const float cell = xi[4];
    const bool  cm   = (cell == 1.0f);
    const float cmf  = cm ? 1.f : 0.f;
    const float o0 = tanhf(acc0 + b20) * cmf;
    const float o1 = tanhf(acc1 + b21) * cmf;
    const float v0x = xi[2], v0y = xi[3];
    float vx = v0x + o0 * ACC_SCALE_;
    float vy = v0y + o1 * ACC_SCALE_;
    const float sp = sqrtf(vx * vx + vy * vy);
    const float sc = fminf(1.f, MAX_VEL_ / (sp + 1e-12f));
    vx *= sc; vy *= sc;
    if (!cm) { vx = v0x; vy = v0y; }
    float npx = pix, npy = piy;
    if (cm) {
      npx = pix + vx; npx -= floorf(npx);         // (pos+vel) mod 1.0
      npy = piy + vy; npy -= floorf(npy);
    }
    const float* nu = noise_u + (size_t)gp * 3;
    const float nm = (cm && nu[2] > 0.5f) ? 1.f : 0.f;
    vx += (nu[0] * 2.f - 1.f) * NOISE_ * nm;
    vy += (nu[1] * 2.f - 1.f) * NOISE_ * nm;
    const bool dead     = cm && (deg < 1);
    const bool consumed = (!cm) && (deg >= 1);
    const float keep = (dead || consumed) ? 0.f : 1.f;

    if (lane < C_) {
      float val;
      switch (lane) {
        case 0:  val = npx; break;
        case 1:  val = npy; break;
        case 2:  val = vx;  break;
        case 3:  val = vy;  break;
        default: val = xi[lane]; break;           // channels 4..9 unchanged
      }
      out[(size_t)gp * C_ + lane] = val * keep;
    }
  }
}

// ---------------------------------------------------------------------------
extern "C" void kernel_launch(void* const* d_in, const int* in_sizes, int n_in,
                              void* d_out, int out_size, void* d_ws, size_t ws_size,
                              hipStream_t stream) {
  (void)in_sizes; (void)n_in; (void)out_size; (void)ws_size;
  const float* x   = (const float*)d_in[0];
  const float* W1  = (const float*)d_in[1];
  const float* b1  = (const float*)d_in[2];
  const float* W2  = (const float*)d_in[3];
  const float* b2  = (const float*)d_in[4];
  const float* nu  = (const float*)d_in[5];
  float* out = (float*)d_out;

  char* ws = (char*)d_ws;
  int*   nbr_idx  = (int*)ws;                                    // B*M*K int
  float* nbr_dist = (float*)(ws + (size_t)B_ * M_ * K_ * 4);     // B*M*K f32
  int*   nbr_cnt  = (int*)(ws + (size_t)B_ * M_ * K_ * 8);       // B*M   int

  knn_kernel<<<(B_ * M_) / 128, 128, 0, stream>>>(x, nbr_idx, nbr_dist, nbr_cnt);
  gnca_mlp_kernel<<<(B_ * M_) / 16, 128, 0, stream>>>(
      x, W1, b1, W2, b2, nu, nbr_idx, nbr_dist, nbr_cnt, out);
}
